// CausalSelfAttention_45483703664859
// MI455X (gfx1250) — matrix operand order repaired and checked by reading, not compiled, not run on hardware
//
#include <hip/hip_runtime.h>
#include <hip/hip_bf16.h>
#include <math.h>

typedef __attribute__((ext_vector_type(16))) _Float16 v16h;
typedef __attribute__((ext_vector_type(8)))  _Float16 v8h;
typedef __attribute__((ext_vector_type(8)))  float    v8f;

#define BATCH 2
#define SEQ   2048
#define DMODEL 768
#define NHEAD 12
#define HD    64
#define MROWS (BATCH * SEQ)      // 4096
#define NO1   (3 * DMODEL)       // 2304

// ---------------------------------------------------------------------------
// Fragment loaders (wave32, WMMA f16 16x16x32 layouts per CDNA5 ISA)
// ---------------------------------------------------------------------------

// A-matrix 16x32 f16: lane L -> row m0+(L%16); halves: e0..7 -> K=kb+h*8+e,
// e8..15 -> K=kb+16+h*8+(e-8), h = L/16. Two 16B contiguous chunks.
__device__ __forceinline__ v16h load_a_frag(const _Float16* base, int ld,
                                            int m0, int kb, int lane) {
  int m = m0 + (lane & 15);
  int h = (lane >> 4) & 1;
  const _Float16* p = base + (size_t)m * ld + kb + h * 8;
  union { v16h v; v8h c[2]; } u;
  u.c[0] = *(const v8h*)(p);
  u.c[1] = *(const v8h*)(p + 16);
  return u.v;
}

// B-matrix 32x16 f16: lane L -> column n0+(L%16); e0..7 -> K=kb+h*8+e, e8..15 -> K=kb+16+h*8+(e-8).
// Operand buffer must be K-contiguous per column (ld = column stride).
__device__ __forceinline__ v16h load_b_frag(const _Float16* base, int ld,
                                            int n0, int kb, int lane) {
  int n = n0 + (lane & 15);
  int h = (lane >> 4) & 1;
  const _Float16* p = base + (size_t)n * ld + kb + h * 8;
  union { v16h v; v8h c[2]; } u;
  u.c[0] = *(const v8h*)(p);
  u.c[1] = *(const v8h*)(p + 16);
  return u.v;
}

__device__ __forceinline__ v8f wmma16(v16h a, v16h b, v8f c) {
  return __builtin_amdgcn_wmma_f32_16x16x32_f16(
      /*neg_a=*/false, a, /*neg_b=*/false, b,
      /*c_mod=*/(short)0, c, /*reuse_a=*/false, /*reuse_b=*/false);
}

// ---------------------------------------------------------------------------
// Prep kernels: f32 -> f16 convert, and transposing weight convert
// ---------------------------------------------------------------------------
__global__ void cvt_f32_to_f16(const float* __restrict__ src,
                               _Float16* __restrict__ dst, int n) {
  int i = blockIdx.x * 256 + threadIdx.x;
  if (i < n) dst[i] = (_Float16)src[i];
}

// Wt[c*K + k] = (f16) W[k*NO + c]   (K = inner dim, NO = out cols)
__global__ void transpose_cvt(const float* __restrict__ W,
                              _Float16* __restrict__ Wt, int K, int NO) {
  int i = blockIdx.x * 256 + threadIdx.x;
  if (i < K * NO) {
    int c = i / K, k = i - c * K;
    Wt[i] = (_Float16)W[(size_t)k * NO + c];
  }
}

// ---------------------------------------------------------------------------
// Stage 1: kqv = x @ Wkqv + b, scattered into Q,K (row-major [bh][n][64])
// and V^T ([bh][64][n]) f16 buffers. One wave = 32 rows x 64 cols,
// double-buffered A fragments across K-steps.
// ---------------------------------------------------------------------------
__global__ void kqv_gemm(const _Float16* __restrict__ xh,
                         const _Float16* __restrict__ Wt,   // [2304][768]
                         const float* __restrict__ bias,    // [2304]
                         _Float16* __restrict__ Qb,
                         _Float16* __restrict__ Kb,
                         _Float16* __restrict__ Vt) {
  int lane = threadIdx.x & 31;
  int w    = threadIdx.x >> 5;
  int m0   = blockIdx.x * 128 + w * 32;
  int c0   = blockIdx.y * 64;

  v8f acc[8];
#pragma unroll
  for (int t = 0; t < 8; ++t) acc[t] = v8f{};

  v16h a0 = load_a_frag(xh, DMODEL, m0,      0, lane);
  v16h a1 = load_a_frag(xh, DMODEL, m0 + 16, 0, lane);

  for (int kb = 0; kb < DMODEL; kb += 32) {
    v16h na0 = a0, na1 = a1;
    if (kb + 32 < DMODEL) {              // prefetch next K-step's A frags
      na0 = load_a_frag(xh, DMODEL, m0,      kb + 32, lane);
      na1 = load_a_frag(xh, DMODEL, m0 + 16, kb + 32, lane);
    }
#pragma unroll
    for (int t = 0; t < 4; ++t) {
      v16h b = load_b_frag(Wt, DMODEL, c0 + t * 16, kb, lane);
      acc[t]     = wmma16(a0, b, acc[t]);
      acc[4 + t] = wmma16(a1, b, acc[4 + t]);
    }
    a0 = na0; a1 = na1;
  }

#pragma unroll
  for (int half = 0; half < 2; ++half) {
    int rbase = m0 + half * 16 + ((lane >> 4) & 1) * 8;
#pragma unroll
    for (int t = 0; t < 4; ++t) {
      int colg = c0 + t * 16 + (lane & 15);
      float bv = bias[colg];
      int ch = colg / DMODEL;            // 0=k, 1=q, 2=v (reference order!)
      int d  = colg - ch * DMODEL;
      int head = d >> 6, hdi = d & 63;
#pragma unroll
      for (int r = 0; r < 8; ++r) {
        int mg = rbase + r;
        int b_ = mg >> 11, nn = mg & (SEQ - 1);
        int bh = b_ * NHEAD + head;
        _Float16 hv = (_Float16)(acc[half * 4 + t][r] + bv);
        if (ch == 0)      Kb[((size_t)bh * SEQ + nn) * HD + hdi] = hv;
        else if (ch == 1) Qb[((size_t)bh * SEQ + nn) * HD + hdi] = hv;
        else              Vt[((size_t)bh * HD + hdi) * SEQ + nn] = hv;
      }
    }
  }
}

// ---------------------------------------------------------------------------
// Stage 2: flash-style causal attention. grid = (SEQ/64, B*H), block = 128.
// Each wave owns 16 query rows; 32 keys/iter with online softmax.
// V frags issued early; next-iter K frags prefetched behind the S WMMAs so
// the exp/shuffle VALU section hides their latency.
// ---------------------------------------------------------------------------
__global__ void attn_kernel(const _Float16* __restrict__ Qb,
                            const _Float16* __restrict__ Kb,
                            const _Float16* __restrict__ Vt,
                            _Float16* __restrict__ sah) {
  int lane = threadIdx.x & 31;
  int w    = threadIdx.x >> 5;
  int bh   = blockIdx.y;
  int b_   = bh / NHEAD;
  int head = bh - b_ * NHEAD;
  int q0   = blockIdx.x * 64 + w * 16;

  const _Float16* Q = Qb + (size_t)bh * SEQ * HD;
  const _Float16* K = Kb + (size_t)bh * SEQ * HD;
  const _Float16* V = Vt + (size_t)bh * HD * SEQ;

  __shared__ __align__(16) _Float16 Pst[4][16][32];   // per-wave P bounce

  v16h qa0 = load_a_frag(Q, HD, q0, 0, lane);
  v16h qa1 = load_a_frag(Q, HD, q0, 32, lane);

  v8f o[4] = {v8f{}, v8f{}, v8f{}, v8f{}};
  float mrow[8], lrow[8];
#pragma unroll
  for (int r = 0; r < 8; ++r) { mrow[r] = -INFINITY; lrow[r] = 0.0f; }

  int rowb = q0 + ((lane >> 4) & 1) * 8;              // C-layout row base
  const float scale = 0.125f;                         // 1/sqrt(64)
  int nsteps = (q0 + 47) >> 5;                        // keys 0..q0+15 covered

  // preload K fragments for the first key tile pair
  v16h kA0 = load_b_frag(K, HD, 0,  0,  lane);
  v16h kA1 = load_b_frag(K, HD, 0,  32, lane);
  v16h kB0 = load_b_frag(K, HD, 16, 0,  lane);
  v16h kB1 = load_b_frag(K, HD, 16, 32, lane);

  for (int it = 0; it < nsteps; ++it) {
    int j0 = it * 32;

    // V fragments for this iteration: issue early, consumed last
    v16h vb0 = load_b_frag(V, SEQ, 0,  j0, lane);
    v16h vb1 = load_b_frag(V, SEQ, 16, j0, lane);
    v16h vb2 = load_b_frag(V, SEQ, 32, j0, lane);
    v16h vb3 = load_b_frag(V, SEQ, 48, j0, lane);

    // S = Q K^T for 16x32 keys
    v8f s0 = v8f{}, s1 = v8f{};
    s0 = wmma16(qa0, kA0, s0);
    s0 = wmma16(qa1, kA1, s0);
    s1 = wmma16(qa0, kB0, s1);
    s1 = wmma16(qa1, kB1, s1);

    // prefetch next iteration's K fragments (hidden behind softmax VALU)
    if (it + 1 < nsteps) {
      int j1 = j0 + 32;
      kA0 = load_b_frag(K, HD, j1,      0,  lane);
      kA1 = load_b_frag(K, HD, j1,      32, lane);
      kB0 = load_b_frag(K, HD, j1 + 16, 0,  lane);
      kB1 = load_b_frag(K, HD, j1 + 16, 32, lane);
    }

    int c0 = j0 + (lane & 15), c1 = c0 + 16;
    float pm[8];
#pragma unroll
    for (int r = 0; r < 8; ++r) {
      int rg = rowb + r;
      float a0 = (c0 <= rg) ? s0[r] * scale : -INFINITY;
      float a1 = (c1 <= rg) ? s1[r] * scale : -INFINITY;
      s0[r] = a0; s1[r] = a1;
      pm[r] = fmaxf(a0, a1);
    }
#pragma unroll
    for (int r = 0; r < 8; ++r) {
      pm[r] = fmaxf(pm[r], __shfl_xor(pm[r], 1, 32));
      pm[r] = fmaxf(pm[r], __shfl_xor(pm[r], 2, 32));
      pm[r] = fmaxf(pm[r], __shfl_xor(pm[r], 4, 32));
      pm[r] = fmaxf(pm[r], __shfl_xor(pm[r], 8, 32));
    }
    float alpha[8], rsum[8];
#pragma unroll
    for (int r = 0; r < 8; ++r) {
      float mn = fmaxf(mrow[r], pm[r]);
      alpha[r] = __expf(mrow[r] - mn);
      mrow[r]  = mn;
      float p0 = __expf(s0[r] - mn);
      float p1 = __expf(s1[r] - mn);
      s0[r] = p0; s1[r] = p1;
      rsum[r] = p0 + p1;
    }
#pragma unroll
    for (int r = 0; r < 8; ++r) {
      rsum[r] += __shfl_xor(rsum[r], 1, 32);
      rsum[r] += __shfl_xor(rsum[r], 2, 32);
      rsum[r] += __shfl_xor(rsum[r], 4, 32);
      rsum[r] += __shfl_xor(rsum[r], 8, 32);
      lrow[r] = lrow[r] * alpha[r] + rsum[r];
    }
#pragma unroll
    for (int t = 0; t < 4; ++t)
#pragma unroll
      for (int r = 0; r < 8; ++r) o[t][r] *= alpha[r];

    // C-layout -> A-layout for P via per-wave LDS bounce
    int rl = ((lane >> 4) & 1) * 8;
#pragma unroll
    for (int r = 0; r < 8; ++r) {
      Pst[w][rl + r][lane & 15]        = (_Float16)s0[r];
      Pst[w][rl + r][16 + (lane & 15)] = (_Float16)s1[r];
    }
    asm volatile("s_wait_dscnt 0" ::: "memory");
    v16h pa = load_a_frag(&Pst[w][0][0], 32, 0, 0, lane);

    // O += P (16x32) @ V (32x64)
    o[0] = wmma16(pa, vb0, o[0]);
    o[1] = wmma16(pa, vb1, o[1]);
    o[2] = wmma16(pa, vb2, o[2]);
    o[3] = wmma16(pa, vb3, o[3]);
    asm volatile("" ::: "memory");   // keep LDS reuse ordered across iters
  }

  // normalize and write sa in row-major f16 [B*SEQ][768]
#pragma unroll
  for (int t = 0; t < 4; ++t) {
    int colg = head * HD + t * 16 + (lane & 15);
#pragma unroll
    for (int r = 0; r < 8; ++r) {
      int rg = rowb + r;
      sah[((size_t)(b_ * SEQ + rg)) * DMODEL + colg] =
          (_Float16)(o[t][r] / lrow[r]);
    }
  }
}

// ---------------------------------------------------------------------------
// Stage 3: out = sa @ Wproj + bproj (f32 output), 32x64 per wave
// ---------------------------------------------------------------------------
__global__ void proj_gemm(const _Float16* __restrict__ sah,
                          const _Float16* __restrict__ Wt,   // [768][768]
                          const float* __restrict__ bias,
                          float* __restrict__ out) {
  int lane = threadIdx.x & 31;
  int w    = threadIdx.x >> 5;
  int m0   = blockIdx.x * 128 + w * 32;
  int c0   = blockIdx.y * 64;

  v8f acc[8];
#pragma unroll
  for (int t = 0; t < 8; ++t) acc[t] = v8f{};

  v16h a0 = load_a_frag(sah, DMODEL, m0,      0, lane);
  v16h a1 = load_a_frag(sah, DMODEL, m0 + 16, 0, lane);

  for (int kb = 0; kb < DMODEL; kb += 32) {
    v16h na0 = a0, na1 = a1;
    if (kb + 32 < DMODEL) {
      na0 = load_a_frag(sah, DMODEL, m0,      kb + 32, lane);
      na1 = load_a_frag(sah, DMODEL, m0 + 16, kb + 32, lane);
    }
#pragma unroll
    for (int t = 0; t < 4; ++t) {
      v16h b = load_b_frag(Wt, DMODEL, c0 + t * 16, kb, lane);
      acc[t]     = wmma16(a0, b, acc[t]);
      acc[4 + t] = wmma16(a1, b, acc[4 + t]);
    }
    a0 = na0; a1 = na1;
  }

#pragma unroll
  for (int half = 0; half < 2; ++half) {
    int rbase = m0 + half * 16 + ((lane >> 4) & 1) * 8;
#pragma unroll
    for (int t = 0; t < 4; ++t) {
      int colg = c0 + t * 16 + (lane & 15);
      float bv = bias[colg];
#pragma unroll
      for (int r = 0; r < 8; ++r)
        out[(size_t)(rbase + r) * DMODEL + colg] = acc[half * 4 + t][r] + bv;
    }
  }
}

// ---------------------------------------------------------------------------
extern "C" void kernel_launch(void* const* d_in, const int* in_sizes, int n_in,
                              void* d_out, int out_size, void* d_ws, size_t ws_size,
                              hipStream_t stream) {
  const float* x     = (const float*)d_in[0];
  const float* Wkqv  = (const float*)d_in[1];
  const float* bkqv  = (const float*)d_in[2];
  const float* Wproj = (const float*)d_in[3];
  const float* bproj = (const float*)d_in[4];
  float* out = (float*)d_out;

  char* ws = (char*)d_ws;
  size_t off = 0;
  _Float16* xh    = (_Float16*)(ws + off); off += (size_t)MROWS * DMODEL * 2;
  _Float16* WtKQV = (_Float16*)(ws + off); off += (size_t)NO1 * DMODEL * 2;
  _Float16* WtP   = (_Float16*)(ws + off); off += (size_t)DMODEL * DMODEL * 2;
  _Float16* Qb    = (_Float16*)(ws + off); off += (size_t)BATCH * NHEAD * SEQ * HD * 2;
  _Float16* Kb    = (_Float16*)(ws + off); off += (size_t)BATCH * NHEAD * SEQ * HD * 2;
  _Float16* Vt    = (_Float16*)(ws + off); off += (size_t)BATCH * NHEAD * HD * SEQ * 2;
  _Float16* sah   = (_Float16*)(ws + off); off += (size_t)MROWS * DMODEL * 2;
  (void)ws_size; (void)in_sizes; (void)n_in; (void)out_size;

  {
    int n = MROWS * DMODEL;
    cvt_f32_to_f16<<<(n + 255) / 256, 256, 0, stream>>>(x, xh, n);
  }
  {
    int n = DMODEL * NO1;
    transpose_cvt<<<(n + 255) / 256, 256, 0, stream>>>(Wkqv, WtKQV, DMODEL, NO1);
  }
  {
    int n = DMODEL * DMODEL;
    transpose_cvt<<<(n + 255) / 256, 256, 0, stream>>>(Wproj, WtP, DMODEL, DMODEL);
  }

  kqv_gemm<<<dim3(MROWS / 128, NO1 / 64), 128, 0, stream>>>(
      xh, WtKQV, bkqv, Qb, Kb, Vt);

  attn_kernel<<<dim3(SEQ / 64, BATCH * NHEAD), 128, 0, stream>>>(
      Qb, Kb, Vt, sah);

  proj_gemm<<<dim3(MROWS / 128, DMODEL / 64), 128, 0, stream>>>(
      sah, WtP, bproj, out);
}
